// BiAttention_43181601194275
// MI455X (gfx1250) — matrix operand order repaired and checked by reading, not compiled, not run on hardware
//
#include <hip/hip_runtime.h>
#include <hip/hip_bf16.h>
#include <math.h>

typedef __bf16 bf16;
typedef __attribute__((ext_vector_type(16))) __bf16 v16bf;
typedef __attribute__((ext_vector_type(8)))  __bf16 v8bf;
typedef __attribute__((ext_vector_type(8)))  float   v8f;

static constexpr int   Bb   = 16;
static constexpr int   Nn   = 2048;
static constexpr int   Mm   = 2048;
static constexpr int   Dd   = 1024;
static constexpr float NEGV = -2.0e20f;

#define DEVI __device__ __forceinline__

DEVI v8f wmma_bf16(v16bf a, v16bf b, v8f c) {
  // D = A(16x32 bf16) * B(32x16 bf16) + C(16x16 f32)
  return __builtin_amdgcn_wmma_f32_16x16x32_bf16(false, a, false, b, (short)0, c, false, false);
}

DEVI v16bf mk16(v8bf lo, v8bf hi) {
  return __builtin_shufflevector(lo, hi, 0, 1, 2, 3, 4, 5, 6, 7, 8, 9, 10, 11, 12, 13, 14, 15);
}

// A-fragment (16x32 bf16) from a bf16 row-major row; sa = (lane<16)?0:8
DEVI v16bf load_a_bf16(const bf16* row, int k, int sa) {
  v8bf lo = *(const v8bf*)(row + k + 2 * sa);
  v8bf hi = *(const v8bf*)(row + k + 2 * sa + 8);
  return mk16(lo, hi);
}

// A-fragment from an f32 row (convert to bf16 on the fly)
DEVI v16bf load_a_f32(const float* row, int k, int sa) {
  const float* p0 = row + k + 2 * sa;
  const float* p1 = row + k + 2 * sa + 8;
  v16bf a;
#pragma unroll
  for (int i = 0; i < 8; ++i) { a[i] = (bf16)p0[i]; a[i + 8] = (bf16)p1[i]; }
  return a;
}

// ---------------------------------------------------------------- utilities

__global__ void cvt_f32_bf16(const float* __restrict__ in, bf16* __restrict__ out, int n) {
  int i = blockIdx.x * blockDim.x + threadIdx.x;
  if (i < n) out[i] = (bf16)in[i];
}

// X: [B, R, D] f32  ->  XT: [B, D, R] bf16   (LDS-tiled transpose, coalesced both ways)
__global__ void transpose_cvt(const float* __restrict__ X, bf16* __restrict__ XT, int R, int D) {
  __shared__ float t[32][33];
  int b  = blockIdx.z;
  int d0 = blockIdx.x * 32, r0 = blockIdx.y * 32;
  const float* Xb  = X  + (size_t)b * R * D;
  bf16*        XTb = XT + (size_t)b * D * R;
  for (int j = threadIdx.y; j < 32; j += blockDim.y)
    t[j][threadIdx.x] = Xb[(size_t)(r0 + j) * D + d0 + threadIdx.x];
  __syncthreads();
  for (int j = threadIdx.y; j < 32; j += blockDim.y)
    XTb[(size_t)(d0 + j) * R + r0 + threadIdx.x] = (bf16)t[threadIdx.x][j];
}

// ------------------------------------------------------- projection (x@W^T+b, ReLU)
// One wave per 32(row) x 64(col) output tile; 8 WMMAs per K-step.
__global__ void proj_kernel(const float* __restrict__ X, const bf16* __restrict__ W,
                            const float* __restrict__ bias, bf16* __restrict__ Out, int rows) {
  const int D = Dd;
  int lane = threadIdx.x & 31;
  int wave = (blockIdx.x * blockDim.x + threadIdx.x) >> 5;
  int nct  = D / 64;
  int rowTile = wave / nct, colTile = wave % nct;
  int n0 = rowTile * 32, e0 = colTile * 64;
  if (n0 >= rows) return;

  const float* arow0 = X + (size_t)(n0 + (lane & 15)) * D;
  const float* arow1 = arow0 + (size_t)16 * D;
  const bf16*  bcol[4];
#pragma unroll
  for (int t = 0; t < 4; ++t) bcol[t] = W + (size_t)(e0 + 16 * t + (lane & 15)) * D;

  int sa = (lane < 16) ? 0 : 8;
  int sb = (lane < 16) ? 0 : 16;
  v8f acc[8] = {};
  for (int k = 0; k < D; k += 32) {
    v16bf a0 = load_a_f32(arow0, k, sa);
    v16bf a1 = load_a_f32(arow1, k, sa);
    v16bf bv[4];
#pragma unroll
    for (int t = 0; t < 4; ++t) bv[t] = *(const v16bf*)(bcol[t] + k + sb);
#pragma unroll
    for (int t = 0; t < 4; ++t) {
      acc[t]     = wmma_bf16(a0, bv[t], acc[t]);
      acc[4 + t] = wmma_bf16(a1, bv[t], acc[4 + t]);
    }
  }
  int rb = (lane < 16) ? 0 : 8;
#pragma unroll
  for (int t = 0; t < 4; ++t) {
    int col = e0 + 16 * t + (lane & 15);
    float bv = bias[col];
#pragma unroll
    for (int v = 0; v < 8; ++v) {
      float r0v = acc[t][v] + bv;
      float r1v = acc[4 + t][v] + bv;
      r0v = r0v > 0.f ? r0v : 0.f;
      r1v = r1v > 0.f ? r1v : 0.f;
      Out[(size_t)(n0 + rb + v) * D + col]      = (bf16)r0v;
      Out[(size_t)(n0 + 16 + rb + v) * D + col] = (bf16)r1v;
    }
  }
}

// ------------------------------------------------------- sim = x1p @ x2p^T, masked
// Block (8 waves) computes 128(n) x 64(m); stages in LDS; writes sim (x2 mask) and
// simT (both masks, == reference sim2) coalesced.
__global__ void sim_kernel(const bf16* __restrict__ X1p, const bf16* __restrict__ X2p,
                           const unsigned char* __restrict__ mask1,
                           const unsigned char* __restrict__ mask2,
                           bf16* __restrict__ Sim, bf16* __restrict__ SimT) {
  const int D = Dd, N = Nn, M = Mm;
  __shared__ float S[128][65];
  int b = blockIdx.z, m0 = blockIdx.x * 64, n0 = blockIdx.y * 128;
  int lane = threadIdx.x & 31, w = threadIdx.x >> 5;

  const bf16* arow = X1p + (size_t)b * N * D + (size_t)(n0 + w * 16 + (lane & 15)) * D;
  const bf16* bcol[4];
#pragma unroll
  for (int t = 0; t < 4; ++t)
    bcol[t] = X2p + (size_t)b * M * D + (size_t)(m0 + 16 * t + (lane & 15)) * D;

  int sa = (lane < 16) ? 0 : 8, sb = (lane < 16) ? 0 : 16;
  v8f acc[4] = {};
  for (int k = 0; k < D; k += 32) {
    __builtin_prefetch(arow + k + 128, 0, 1);
    v16bf a = load_a_bf16(arow, k, sa);
    v16bf bv[4];
#pragma unroll
    for (int t = 0; t < 4; ++t) bv[t] = *(const v16bf*)(bcol[t] + k + sb);
#pragma unroll
    for (int t = 0; t < 4; ++t) acc[t] = wmma_bf16(a, bv[t], acc[t]);
  }
  int rb = (lane < 16) ? 0 : 8;
#pragma unroll
  for (int t = 0; t < 4; ++t)
#pragma unroll
    for (int v = 0; v < 8; ++v)
      S[w * 16 + rb + v][16 * t + (lane & 15)] = acc[t][v];
  __syncthreads();

  const unsigned char* m1 = mask1 + (size_t)b * N + n0;
  const unsigned char* m2 = mask2 + (size_t)b * M + m0;
  for (int idx = threadIdx.x; idx < 128 * 64; idx += blockDim.x) {
    int n = idx >> 6, m = idx & 63;
    float v = m2[m] ? NEGV : S[n][m];
    Sim[(size_t)b * N * M + (size_t)(n0 + n) * M + m0 + m] = (bf16)v;
  }
  for (int idx = threadIdx.x; idx < 64 * 128; idx += blockDim.x) {
    int m = idx >> 7, n = idx & 127;
    float v = (m2[m] || m1[n]) ? NEGV : S[n][m];
    SimT[(size_t)b * M * N + (size_t)(m0 + m) * N + n0 + n] = (bf16)v;
  }
}

// ------------------------------- softmax stats + in-place exp (logits -> P)
// Pass 1: row max. Pass 2: row sum of exp. Pass 3: overwrite row with
// P = bf16(exp(x - max)); 1/sum applied later on f32 accumulators.
__global__ void stats_kernel(bf16* __restrict__ S, float* __restrict__ inv, int L) {
  __shared__ float red[256];
  size_t row = blockIdx.x;
  bf16* p = S + row * L;
  float m = -INFINITY;
  for (int i = threadIdx.x; i < L; i += blockDim.x) m = fmaxf(m, (float)p[i]);
  red[threadIdx.x] = m; __syncthreads();
  for (int s = 128; s > 0; s >>= 1) {
    if ((int)threadIdx.x < s) red[threadIdx.x] = fmaxf(red[threadIdx.x], red[threadIdx.x + s]);
    __syncthreads();
  }
  m = red[0]; __syncthreads();
  float sum = 0.f;
  for (int i = threadIdx.x; i < L; i += blockDim.x) sum += __expf((float)p[i] - m);
  red[threadIdx.x] = sum; __syncthreads();
  for (int s = 128; s > 0; s >>= 1) {
    if ((int)threadIdx.x < s) red[threadIdx.x] += red[threadIdx.x + s];
    __syncthreads();
  }
  if (threadIdx.x == 0) inv[row] = 1.f / red[0];
  // pass 3: exponentiate in place (each row owned by exactly one block)
  for (int i = threadIdx.x; i < L; i += blockDim.x)
    p[i] = (bf16)__expf((float)p[i] - m);
}

// ------------------------------------------------------- attn = P @ V (pure GEMM)
// P: [B, R, L] bf16 un-normalized softmax weights; VT: [B, D, L] bf16;
// Out: [B, R, D] f32, rows scaled by 1/rowsum. 32(row) x 64(col) per wave.
__global__ void attn_kernel(const bf16* __restrict__ P, const bf16* __restrict__ VT,
                            const float* __restrict__ rinv,
                            float* __restrict__ Out, int R, int L) {
  const int D = Dd;
  int b = blockIdx.z, lane = threadIdx.x & 31, w = threadIdx.x >> 5;
  int r0 = (blockIdx.y * 8 + w) * 32, e0 = blockIdx.x * 64;

  const bf16*  Pb  = P    + (size_t)b * R * L;
  const bf16*  Vb  = VT   + (size_t)b * D * L;
  const float* ivb = rinv + (size_t)b * R;

  const bf16* arow0 = Pb + (size_t)(r0 + (lane & 15)) * L;
  const bf16* arow1 = arow0 + (size_t)16 * L;
  const bf16* bcol[4];
#pragma unroll
  for (int t = 0; t < 4; ++t) bcol[t] = Vb + (size_t)(e0 + 16 * t + (lane & 15)) * L;

  int sa = (lane < 16) ? 0 : 8, sb = (lane < 16) ? 0 : 16;
  v8f acc[8] = {};
  for (int k = 0; k < L; k += 32) {
    __builtin_prefetch(arow0 + k + 128, 0, 1);
    v16bf a0 = load_a_bf16(arow0, k, sa);
    v16bf a1 = load_a_bf16(arow1, k, sa);
    v16bf bv[4];
#pragma unroll
    for (int t = 0; t < 4; ++t) bv[t] = *(const v16bf*)(bcol[t] + k + sb);
#pragma unroll
    for (int t = 0; t < 4; ++t) {
      acc[t]     = wmma_bf16(a0, bv[t], acc[t]);
      acc[4 + t] = wmma_bf16(a1, bv[t], acc[4 + t]);
    }
  }
  int rb = (lane < 16) ? 0 : 8;
  float* Ob = Out + (size_t)b * R * D;
#pragma unroll
  for (int v = 0; v < 8; ++v) {
    float sc0 = ivb[r0 + rb + v];
    float sc1 = ivb[r0 + 16 + rb + v];
#pragma unroll
    for (int t = 0; t < 4; ++t) {
      int col = e0 + 16 * t + (lane & 15);
      Ob[(size_t)(r0 + rb + v) * D + col]      = acc[t][v] * sc0;
      Ob[(size_t)(r0 + 16 + rb + v) * D + col] = acc[4 + t][v] * sc1;
    }
  }
}

// ---------------------------------------------------------------- launch

extern "C" void kernel_launch(void* const* d_in, const int* in_sizes, int n_in,
                              void* d_out, int out_size, void* d_ws, size_t ws_size,
                              hipStream_t stream) {
  (void)in_sizes; (void)n_in; (void)out_size; (void)ws_size;
  const float*         x1  = (const float*)d_in[0];
  const unsigned char* x1m = (const unsigned char*)d_in[1];
  const float*         x2  = (const float*)d_in[2];
  const unsigned char* x2m = (const unsigned char*)d_in[3];
  const float*         W1  = (const float*)d_in[4];
  const float*         b1  = (const float*)d_in[5];
  const float*         W2  = (const float*)d_in[6];
  const float*         b2  = (const float*)d_in[7];

  float* attn_a = (float*)d_out;
  float* attn_b = attn_a + (size_t)Bb * Nn * Dd;

  char*  ws  = (char*)d_ws;
  size_t off = 0;
  auto alloc = [&](size_t bytes) {
    char* p = ws + off; off += (bytes + 255) & ~(size_t)255; return p;
  };
  bf16*  x1p  = (bf16*)alloc((size_t)Bb * Nn * Dd * 2);
  bf16*  x2p  = (bf16*)alloc((size_t)Bb * Mm * Dd * 2);
  bf16*  x1T  = (bf16*)alloc((size_t)Bb * Dd * Nn * 2);
  bf16*  x2T  = (bf16*)alloc((size_t)Bb * Dd * Mm * 2);
  bf16*  sim  = (bf16*)alloc((size_t)Bb * Nn * Mm * 2);
  bf16*  simT = (bf16*)alloc((size_t)Bb * Mm * Nn * 2);
  bf16*  W1b  = (bf16*)alloc((size_t)Dd * Dd * 2);
  bf16*  W2b  = (bf16*)alloc((size_t)Dd * Dd * 2);
  float* riv  = (float*)alloc((size_t)Bb * Nn * 4);
  float* civ  = (float*)alloc((size_t)Bb * Mm * 4);

  // 1. weights -> bf16
  int wn = Dd * Dd;
  cvt_f32_bf16<<<wn / 256, 256, 0, stream>>>(W1, W1b, wn);
  cvt_f32_bf16<<<wn / 256, 256, 0, stream>>>(W2, W2b, wn);

  // 2. x1 -> x1T, x2 -> x2T (bf16 transposed copies for attention B operands)
  transpose_cvt<<<dim3(Dd / 32, Nn / 32, Bb), dim3(32, 8), 0, stream>>>(x1, x1T, Nn, Dd);
  transpose_cvt<<<dim3(Dd / 32, Mm / 32, Bb), dim3(32, 8), 0, stream>>>(x2, x2T, Mm, Dd);

  // 3. projections (bf16 WMMA GEMM + bias + ReLU)
  {
    int rows  = Bb * Nn;
    int waves = (rows / 32) * (Dd / 64);
    proj_kernel<<<(waves + 7) / 8, 256, 0, stream>>>(x1, W1b, b1, x1p, rows);
    rows  = Bb * Mm;
    waves = (rows / 32) * (Dd / 64);
    proj_kernel<<<(waves + 7) / 8, 256, 0, stream>>>(x2, W2b, b2, x2p, rows);
  }

  // 4. similarity (+ masked transpose)
  sim_kernel<<<dim3(Mm / 64, Nn / 128, Bb), 256, 0, stream>>>(x1p, x2p, x1m, x2m, sim, simT);

  // 5. softmax stats + in-place exp (sim/simT become un-normalized P matrices)
  stats_kernel<<<Bb * Nn, 256, 0, stream>>>(sim, riv, Mm);
  stats_kernel<<<Bb * Mm, 256, 0, stream>>>(simT, civ, Nn);

  // 6. attention outputs (pure bf16 WMMA GEMMs, row-scaled epilogue)
  attn_kernel<<<dim3(Dd / 64, Nn / 256, Bb), 256, 0, stream>>>(sim, x2T, riv, attn_a, Nn, Mm);
  attn_kernel<<<dim3(Dd / 64, Mm / 256, Bb), 256, 0, stream>>>(simT, x1T, civ, attn_b, Mm, Nn);
}
